// TERMGVPEncoder_79817672229197
// MI455X (gfx1250) — matrix operand order repaired and checked by reading, not compiled, not run on hardware
//
#include <hip/hip_runtime.h>

// ---------------------------------------------------------------------------
// TERMGVPEncoder forward for gfx1250 (MI455X). All GEMMs on
// v_wmma_f32_16x16x32_bf16 (bf16 in, f32 accumulate), wave32.
// Templated compile-time dims -> branch-free, vectorized fragment loads.
// ---------------------------------------------------------------------------

typedef __attribute__((ext_vector_type(16))) __bf16 v16bf;
typedef __attribute__((ext_vector_type(8)))  float  v8f;

#define NSC   100
#define B_    4
#define T_    50
#define N_    25
#define K_    25
#define NNODE (B_*T_*N_)      // 5000
#define NEDGE (B_*T_*N_*K_)   // 125000
#define CH    148             // 3*NV + NS

// ---------------- fragment element fillers ---------------------------------

// Fill 8 consecutive elements (jg half of an A fragment) from 8 consecutive
// floats at p (16B aligned) -> two b128 loads + pack converts.
__device__ __forceinline__ void fill8(v16bf& f, int jg, const float* p) {
  const float4* q = (const float4*)p;
  float4 x = q[0], y = q[1];
  int b = 8 * jg;
  f[b + 0] = (__bf16)x.x; f[b + 1] = (__bf16)x.y;
  f[b + 2] = (__bf16)x.z; f[b + 3] = (__bf16)x.w;
  f[b + 4] = (__bf16)y.x; f[b + 5] = (__bf16)y.y;
  f[b + 6] = (__bf16)y.z; f[b + 7] = (__bf16)y.w;
}

__device__ __forceinline__ void zero8(v16bf& f, int jg) {
  int b = 8 * jg;
#pragma unroll
  for (int t = 0; t < 8; ++t) f[b + t] = (__bf16)0.0f;
}

// B fragment (32x16 bf16) from row-major W[k][n]; branch-free (clamped
// addresses + value selects). lanes 0-15: K=k0+{0..7,16..23}, lanes 16-31: +8.
__device__ __forceinline__ v16bf bfrag(const float* __restrict__ W, int ldw,
                                       int ncols, int k0, int kmax,
                                       int n0, int m, int half) {
  int n = n0 + m;
  if (n >= ncols) n = ncols - 1;       // clamped; those C columns never stored
  int kb = k0 + half * 8;
  v16bf f;
#pragma unroll
  for (int t = 0; t < 16; ++t) {
    int k = kb + t + (t >= 8 ? 8 : 0);
    int kc = (k < kmax) ? k : (kmax - 1);
    float v = W[(size_t)kc * ldw + n];
    f[t] = (__bf16)((k < kmax) ? v : 0.0f);
  }
  return f;
}

// ---------------- generic GVP kernel (compile-time dims) -------------------

template <int VI, int H, int VO, int SI, int SO, int NLS, int NLV, int GATHER>
__launch_bounds__(128)
__global__ void gvp_kernel(const float* __restrict__ X, int ldx,
                           const float* __restrict__ hV,
                           const float* __restrict__ hE,
                           const int* __restrict__ idxI,
                           const int* __restrict__ idxJ,
                           const float* __restrict__ wh,
                           const float* __restrict__ wsm,
                           const float* __restrict__ wv,
                           const float* __restrict__ bias,
                           float* __restrict__ Y, int ldy, int R) {
  constexpr int KTOT = SI + H;
  constexpr int KBV  = (VI + 31) / 32;
  constexpr int KBS  = (KTOT + 31) / 32;
  constexpr int KBH  = (H + 31) / 32;
  constexpr int NTH  = H / 16;
  constexpr int NTS  = (SO + 15) / 16;
  constexpr int NTO  = VO / 16;

  __shared__ float smem[4 * 3072];     // per wave: vh[3][16][48] + vn[16][48]
  const int lane = threadIdx.x & 31;
  const int wave = threadIdx.x >> 5;
  const int half = lane >> 4;
  const int m    = lane & 15;
  float* vh    = smem + wave * 3072;
  float* vn    = vh + 2304;
  float* vnRow = vn + m * 48;

  int tiles = (R + 15) >> 4;
  int tile  = blockIdx.x * 4 + wave;
  if (tile >= tiles) tile = tiles - 1;            // duplicate tail (benign)
  const int r0   = tile * 16;
  const int arow = min(r0 + m, R - 1);            // this lane's A row

  size_t rI = 0, rJ = 0, rE = 0;
  const float* Xrow = nullptr;
  if (GATHER) {
    rI = (size_t)idxI[arow] * CH;
    rJ = (size_t)idxJ[arow] * CH;
    rE = (size_t)arow * CH;
  } else {
    Xrow = X + (size_t)arow * ldx;
  }

  // ---- phase 1: vh[c] = v[c] @ wh ; vn = sqrt(clip(sum_c vh^2)) ----
  for (int nt = 0; nt < NTH; ++nt) {
    v8f accs[3];
#pragma unroll
    for (int c = 0; c < 3; ++c) {
      v8f acc = {};
#pragma unroll
      for (int kb = 0; kb < KBV; ++kb) {
        v16bf af;
#pragma unroll
        for (int jg = 0; jg < 2; ++jg) {
          int blk = kb * 32 + jg * 16;            // compile-time constant
          if (blk < VI) {                         // VI%16==0 -> all-or-nothing
            const float* p;
            if (GATHER) {                         // VI==48: [v_i|v_j|v_e]
              int col = c * 16 + half * 8;
              p = (blk == 0)  ? hV + rI + col
                : (blk == 16) ? hV + rJ + col
                              : hE + rE + col;
            } else {
              p = Xrow + c * VI + blk + half * 8;
            }
            fill8(af, jg, p);
          } else {
            zero8(af, jg);
          }
        }
        v16bf bf = bfrag(wh, H, H, kb * 32, VI, nt * 16, m, half);
        acc = __builtin_amdgcn_wmma_f32_16x16x32_bf16(false, af, false, bf,
                                                      (short)0, acc, false, false);
      }
      accs[c] = acc;
      int n = nt * 16 + m;       // C layout: N = lane&15, M = j + 8*half
#pragma unroll
      for (int j = 0; j < 8; ++j)
        vh[(c * 16 + (j + half * 8)) * 48 + n] = acc[j];
    }
    int n = nt * 16 + m;
#pragma unroll
    for (int j = 0; j < 8; ++j) {
      float nsq = accs[0][j] * accs[0][j] + accs[1][j] * accs[1][j] +
                  accs[2][j] * accs[2][j];
      vn[(j + half * 8) * 48 + n] = sqrtf(fmaxf(nsq, 1e-8f));
    }
  }
  // same-wave LDS RAW: DS ops are in-order within a wave; drain DScnt
  asm volatile("s_wait_dscnt 0" ::: "memory");

  // ---- phase 2: so = [s | vn] @ ws + b ; optional relu ----
  for (int nt = 0; nt < NTS; ++nt) {
    v8f acc = {};
#pragma unroll
    for (int kb = 0; kb < KBS; ++kb) {
      v16bf af;
#pragma unroll
      for (int jg = 0; jg < 2; ++jg) {
        int blk = kb * 32 + jg * 16;              // compile-time constant
        if (blk >= KTOT) {
          zero8(af, jg);
        } else if (!GATHER && blk + 16 <= SI) {   // fully in dense scalar region
          fill8(af, jg, Xrow + 3 * VI + blk + half * 8);
        } else if (GATHER && blk + 16 <= 100) {   // fully in s_i segment
          fill8(af, jg, hV + rI + 48 + blk + half * 8);
        } else {                                  // boundary: clamped selects
#pragma unroll
          for (int t = 0; t < 8; ++t) {
            int k = blk + half * 8 + t;
            int kv = k - SI; kv = kv < 0 ? 0 : (kv >= H ? H - 1 : kv);
            float v;
            if (GATHER) {                         // SI==300: [s_i|s_j|s_e]
              const float* p = (k < 100) ? hV + rI + 48 + k
                             : (k < 200) ? hV + rJ + 48 + (k - 100)
                                         : hE + rE + 48 + min(k - 200, 99);
              v = (k < SI) ? *p : vnRow[kv];
            } else {
              v = (k < SI) ? Xrow[3 * VI + min(k, SI - 1)] : vnRow[kv];
            }
            if (k >= KTOT) v = 0.0f;
            af[8 * jg + t] = (__bf16)v;
          }
        }
      }
      v16bf bf = bfrag(wsm, SO, SO, kb * 32, KTOT, nt * 16, m, half);
      acc = __builtin_amdgcn_wmma_f32_16x16x32_bf16(false, af, false, bf,
                                                    (short)0, acc, false, false);
    }
    int n = nt * 16 + m;
    float bb = bias[min(n, SO - 1)];
#pragma unroll
    for (int j = 0; j < 8; ++j) {
      float v = acc[j] + bb;
      if (NLS) v = fmaxf(v, 0.0f);
      int row = r0 + j + half * 8;
      if (row < R && n < SO)
        Y[(size_t)row * ldy + 3 * VO + n] = v;
    }
  }

  // ---- phase 3: vo = vh @ wv ; optional vo *= sigmoid(||vo||) ----
  for (int nt = 0; nt < NTO; ++nt) {
    v8f accs[3];
#pragma unroll
    for (int c = 0; c < 3; ++c) {
      v8f acc = {};
#pragma unroll
      for (int kb = 0; kb < KBH; ++kb) {
        v16bf af;
#pragma unroll
        for (int jg = 0; jg < 2; ++jg) {
          int blk = kb * 32 + jg * 16;            // compile-time constant
          if (blk < H) fill8(af, jg, vh + c * 768 + m * 48 + blk + half * 8);
          else         zero8(af, jg);
        }
        v16bf bf = bfrag(wv, VO, VO, kb * 32, H, nt * 16, m, half);
        acc = __builtin_amdgcn_wmma_f32_16x16x32_bf16(false, af, false, bf,
                                                      (short)0, acc, false, false);
      }
      accs[c] = acc;
    }
    int n = nt * 16 + m;
    float sig[8];
#pragma unroll
    for (int j = 0; j < 8; ++j) {
      if (NLV) {
        float nsq = accs[0][j] * accs[0][j] + accs[1][j] * accs[1][j] +
                    accs[2][j] * accs[2][j];
        float nrm = sqrtf(fmaxf(nsq, 1e-8f));
        sig[j] = 1.0f / (1.0f + expf(-nrm));
      } else {
        sig[j] = 1.0f;
      }
    }
#pragma unroll
    for (int c = 0; c < 3; ++c)
#pragma unroll
      for (int j = 0; j < 8; ++j) {
        int row = r0 + j + half * 8;
        if (row < R)
          Y[(size_t)row * ldy + c * VO + n] = accs[c][j] * sig[j];
      }
  }
}

// ---------------- edge index / attend-mask prep ----------------------------

__global__ void prep_kernel(const int* __restrict__ E_idx,
                            const float* __restrict__ mask,
                            int* __restrict__ idxI, int* __restrict__ idxJ,
                            float* __restrict__ maskA, int NE) {
  int e = blockIdx.x * blockDim.x + threadIdx.x;
  if (e >= NE) return;
  int node = e / K_;
  int bt   = node / N_;
  int j  = bt * N_ + E_idx[e];
  int i0 = bt * N_ + E_idx[node * K_];    // i_idx = E_idx[..., 0] broadcast
  idxJ[e] = j;
  idxI[e] = i0;
  maskA[e] = mask[node] * mask[j];
}

// ---------------- masked K-mean + residual + GVP layer norm ----------------

__global__ void residual_norm_kernel(const float* __restrict__ hVin,
                                     const float* __restrict__ dh,
                                     const float* __restrict__ maskA, int Kn,
                                     const float* __restrict__ gamma,
                                     const float* __restrict__ beta,
                                     const float* __restrict__ mask,
                                     int applyMask, float* __restrict__ out) {
  __shared__ float sx[CH];
  __shared__ float red[16];
  __shared__ float vrms, smu, svar;
  int node = blockIdx.x;
  int t = threadIdx.x;
  for (int ch = t; ch < CH; ch += blockDim.x) {
    float acc = 0.0f;
    for (int k = 0; k < Kn; ++k) {
      float w = maskA ? maskA[node * Kn + k] : 1.0f;
      acc += w * dh[((size_t)node * Kn + k) * CH + ch];
    }
    sx[ch] = hVin[(size_t)node * CH + ch] + acc / (float)Kn;
  }
  __syncthreads();
  if (t < 16) {
    float nsq = 0.0f;
#pragma unroll
    for (int c = 0; c < 3; ++c) { float v = sx[c * 16 + t]; nsq += v * v; }
    red[t] = fmaxf(nsq, 1e-8f);
  }
  __syncthreads();
  if (t == 0) {
    float s = 0.0f;
#pragma unroll
    for (int i = 0; i < 16; ++i) s += red[i];
    vrms = sqrtf(s * (1.0f / 16.0f));
  }
  if (t == 1) {
    float s = 0.0f, s2 = 0.0f;
    for (int i = 0; i < NSC; ++i) { float x = sx[48 + i]; s += x; s2 += x * x; }
    float mu = s * (1.0f / NSC);
    smu = mu; svar = s2 * (1.0f / NSC) - mu * mu;
  }
  __syncthreads();
  float mk  = applyMask ? mask[node] : 1.0f;
  float inv = rsqrtf(svar + 1e-5f);
  for (int ch = t; ch < CH; ch += blockDim.x) {
    float x = sx[ch], y;
    if (ch < 48) y = x / vrms;
    else { int i = ch - 48; y = (x - smu) * inv * gamma[i] + beta[i]; }
    out[(size_t)node * CH + ch] = y * mk;
  }
}

// ---------------- host-side orchestration ----------------------------------

template <int VI, int H, int VO, int SI, int SO, int NLS, int NLV, int GATHER>
static void launch_gvp(const float* X, int ldx, const float* hV,
                       const float* hE, const int* idxI, const int* idxJ,
                       const float* b, const float* wh, const float* wsW,
                       const float* wv, float* Y, int ldy, int R,
                       hipStream_t stream) {
  int blocks = (R + 63) / 64;            // 4 waves/block, 16 rows/wave
  gvp_kernel<VI, H, VO, SI, SO, NLS, NLV, GATHER>
      <<<blocks, 128, 0, stream>>>(X, ldx, hV, hE, idxI, idxJ,
                                   wh, wsW, wv, b, Y, ldy, R);
}

extern "C" void kernel_launch(void* const* d_in, const int* in_sizes, int n_in,
                              void* d_out, int out_size, void* d_ws, size_t ws_size,
                              hipStream_t stream) {
  (void)in_sizes; (void)n_in; (void)out_size;

  const float* V    = (const float*)d_in[0];
  const float* E    = (const float*)d_in[1];
  const int*   Eix  = (const int*)d_in[2];
  const float* mask = (const float*)d_in[3];
  auto P = [&](int i) { return (const float*)d_in[i]; };
  // params flattened pytree (sorted keys): W_e(b,wh,ws,wv)=4..7,
  // W_out=8..11, W_v=12..15, then per layer l (base 16+24l):
  //   W_EV0..2 (b,wh,ws,wv)x3, W_dh0..1 (b,wh,ws,wv)x2, b0,b1,g0,g1

  float* wsf = (float*)d_ws;
  size_t off = 0;
  auto alloc = [&](size_t n) { float* p = wsf + off; off += n; return p; };
  float* hE    = alloc((size_t)NEDGE * CH);
  float* bufA  = alloc((size_t)NEDGE * CH);
  float* bufB  = alloc((size_t)NEDGE * CH);
  float* hV0   = alloc((size_t)NNODE * CH);
  float* hV1   = alloc((size_t)NNODE * CH);
  float* bufN1 = alloc((size_t)NNODE * 496);
  float* bufN2 = alloc((size_t)NNODE * CH);
  float* maskA = alloc(NEDGE);
  int* idxI = (int*)alloc(NEDGE);
  int* idxJ = (int*)alloc(NEDGE);
  if (off * sizeof(float) > ws_size) return;   // scratch too small

  prep_kernel<<<(NEDGE + 255) / 256, 256, 0, stream>>>(Eix, mask, idxI, idxJ,
                                                       maskA, NEDGE);

  // h_V = gvp(V, W_v) ; h_E = gvp(E, W_e)
  launch_gvp<16, 16, 16, 200, 100, 1, 1, 0>(V, 248, nullptr, nullptr, idxI, idxJ,
      P(12), P(13), P(14), P(15), hV0, CH, NNODE, stream);
  launch_gvp<16, 16, 16, 100, 100, 1, 1, 0>(E, CH, nullptr, nullptr, idxI, idxJ,
      P(4), P(5), P(6), P(7), hE, CH, NEDGE, stream);

  for (int l = 0; l < 3; ++l) {
    int bs = 16 + 24 * l;
    // W_EV chain over edges; first GVP fuses the h_EV gather-concat
    launch_gvp<48, 48, 16, 300, 100, 1, 1, 1>(nullptr, 0, hV0, hE, idxI, idxJ,
        P(bs + 0), P(bs + 1), P(bs + 2), P(bs + 3), bufA, CH, NEDGE, stream);
    launch_gvp<16, 16, 16, 100, 100, 1, 1, 0>(bufA, CH, nullptr, nullptr, idxI, idxJ,
        P(bs + 4), P(bs + 5), P(bs + 6), P(bs + 7), bufB, CH, NEDGE, stream);
    launch_gvp<16, 16, 16, 100, 100, 0, 0, 0>(bufB, CH, nullptr, nullptr, idxI, idxJ,
        P(bs + 8), P(bs + 9), P(bs + 10), P(bs + 11), bufA, CH, NEDGE, stream);
    // h_V = LN(h_V + mean_k(mask_attend * dh)) with g0/b0
    residual_norm_kernel<<<NNODE, 64, 0, stream>>>(
        hV0, bufA, maskA, K_, P(bs + 22), P(bs + 20), mask, 0, hV1);
    // W_dh chain over nodes
    launch_gvp<16, 32, 32, 100, 400, 1, 1, 0>(hV1, CH, nullptr, nullptr, idxI, idxJ,
        P(bs + 12), P(bs + 13), P(bs + 14), P(bs + 15), bufN1, 496, NNODE, stream);
    launch_gvp<32, 32, 16, 400, 100, 0, 0, 0>(bufN1, 496, nullptr, nullptr, idxI, idxJ,
        P(bs + 16), P(bs + 17), P(bs + 18), P(bs + 19), bufN2, CH, NNODE, stream);
    // h_V = mask * LN(h_V + dh) with g1/b1
    residual_norm_kernel<<<NNODE, 64, 0, stream>>>(
        hV1, bufN2, nullptr, 1, P(bs + 23), P(bs + 21), mask, 1, hV0);
  }

  // out = gvp(h_V, W_out, no nonlinearities)
  launch_gvp<16, 16, 16, 100, 100, 0, 0, 0>(hV0, CH, nullptr, nullptr, idxI, idxJ,
      P(8), P(9), P(10), P(11), (float*)d_out, CH, NNODE, stream);
}
